// Attention_26173530702697
// MI455X (gfx1250) — matrix operand order repaired and checked by reading, not compiled, not run on hardware
//
#include <hip/hip_runtime.h>
#include <hip/hip_bf16.h>

#define DEV __device__ __forceinline__

typedef __attribute__((ext_vector_type(16))) _Float16 v16h;
typedef __attribute__((ext_vector_type(8)))  _Float16 v8h;
typedef __attribute__((ext_vector_type(8)))  float    v8f;

constexpr int Bc = 8, DIMc = 384, Hc = 32, Wc = 32, HEADSc = 8, DKc = 96;
constexpr int Nc_ = Hc * Wc;          // 1024 tokens
constexpr int HID = 768;              // MLP hidden
constexpr float EPSc = 1e-5f;

// ---------------------------------------------------------------------------
// WMMA fragment helpers (16x16x32 f16 -> f32)
// ---------------------------------------------------------------------------
DEV v16h load_frag_a(const _Float16* __restrict__ A, int lda, int lane) {
  int m  = lane & 15;
  int ko = (lane >> 4) << 3;                 // 0 or 8
  const _Float16* p = A + (long)m * lda + ko;
  v8h lo = *(const v8h*)p;
  v8h hi = *(const v8h*)(p + 16);
  v16h r;
#pragma unroll
  for (int i = 0; i < 8; i++) { r[i] = lo[i]; r[8 + i] = hi[i]; }
  return r;
}

// B fragment (32k x 16 cols) from a row-major [cols][k] tile with leading dim ld
DEV v16h frag_b_ld(const _Float16* lb, int ld, int lane) {
  int n  = lane & 15;
  int ko = (lane >> 4) << 3;                 // 0 or 8
  const _Float16* p = lb + (long)n * ld + ko;
  v8h lo = *(const v8h*)p;
  v8h hi = *(const v8h*)(p + 16);
  v16h r;
#pragma unroll
  for (int i = 0; i < 8; i++) { r[i] = lo[i]; r[8 + i] = hi[i]; }
  return r;
}

DEV v8f wmma_f16(v16h a, v16h b, v8f c) {
  return __builtin_amdgcn_wmma_f32_16x16x32_f16(false, a, false, b, (short)0, c,
                                                false, false);
}

// CDNA5 async global->LDS copy (16B per lane), tracked by ASYNCcnt
DEV void async_copy_b128(unsigned lds_off, const _Float16* gptr) {
  asm volatile("global_load_async_to_lds_b128 %0, %1, off"
               :: "v"(lds_off), "v"(gptr) : "memory");
}
DEV void wait_async0() { asm volatile("s_wait_asynccnt 0x0" ::: "memory"); }

DEV float gelu_exact(float x) { return 0.5f * x * (1.0f + erff(x * 0.70710678118f)); }

// ---------------------------------------------------------------------------
// Generic batched WMMA GEMM:  C[z] = A[z%aMod] * Bt[z/bDiv]^T + bias[z/bDiv]
// block = 256 threads = 8 waves; block tile 256(M) x 32(N); wave tile 32x32.
// B tile (shared by all waves) double-buffered in LDS via async copies;
// A fragments double-buffered in registers (loaded one step ahead).
// mode 1: f32 out + bias + resScale[n]*res      (attention proj / mlp2)
// mode 2: f16 out + bias                        (q/k projections)
// mode 3: f16 TRANSPOSED out + bias (ld = M)    (v projection -> V^T)
// mode 4: f16 out + bias + exact GELU           (mlp1)
// ---------------------------------------------------------------------------
__global__ __launch_bounds__(256)
void gemm_wmma_kernel(const _Float16* __restrict__ A, const _Float16* __restrict__ Bt,
                      const float* __restrict__ bias,
                      float* __restrict__ Cf, _Float16* __restrict__ Ch,
                      const float* __restrict__ res, const float* __restrict__ resScale,
                      int M, int Nc, int K,
                      int aMod, long aStride, int bDiv, long bStride,
                      long cStride, int biasStride, int mode)
{
  __shared__ _Float16 sB[2][32 * 32];          // 2 x 2KB double buffer

  int tid  = threadIdx.x;
  int lane = tid & 31;
  int wv   = tid >> 5;
  int z    = blockIdx.z;
  int m0   = blockIdx.y * 256 + wv * 32;
  int n0   = blockIdx.x * 32;

  const _Float16* Az = A  + (long)(z % aMod) * aStride + (long)m0 * K;
  const _Float16* Bz = Bt + (long)(z / bDiv) * bStride + (long)n0 * K;
  const float* bz    = bias + (long)(z / bDiv) * biasStride;

  // async stager: threads 0..127 move the 32x32-half B tile (2KB, 16B/lane)
  bool stager = (tid < 128);
  int  sn = (tid & 127) >> 2;                  // B-tile row (col of C) 0..31
  int  sc = tid & 3;                           // 16B chunk within row
  const _Float16* gB = Bz + (long)sn * K + sc * 8;
  unsigned ldsDst[2] = {
    (unsigned)(uintptr_t)&sB[0][sn * 32 + sc * 8],
    (unsigned)(uintptr_t)&sB[1][sn * 32 + sc * 8]
  };

  int steps = K / 32;
  if (stager) async_copy_b128(ldsDst[0], gB);

  v8f c00 = {}, c01 = {}, c10 = {}, c11 = {};

  // A fragments one step ahead in registers
  v16h af0 = load_frag_a(Az, K, lane);
  v16h af1 = load_frag_a(Az + (long)16 * K, K, lane);

  for (int i = 0; i < steps; i++) {
    wait_async0();                 // B tile for buffer i&1 has landed (stagers)
    __syncthreads();               // publish to all waves; buffer (i+1)&1 free
    if (stager && (i + 1) < steps)
      async_copy_b128(ldsDst[(i + 1) & 1], gB + (long)(i + 1) * 32);

    // prefetch/preload next A fragments (consumed next iteration)
    v16h an0 = af0, an1 = af1;
    if (i + 1 < steps) {
      int kn = (i + 1) * 32;
      if (kn + 96 < K) __builtin_prefetch(Az + kn + 96, 0, 1);
      an0 = load_frag_a(Az + kn, K, lane);
      an1 = load_frag_a(Az + (long)16 * K + kn, K, lane);
    }

    const _Float16* lb = &sB[i & 1][0];
    v16h bf0 = frag_b_ld(lb, 32, lane);            // cols n0..n0+15
    v16h bf1 = frag_b_ld(lb + 16 * 32, 32, lane);  // cols n0+16..n0+31

    c00 = wmma_f16(af0, bf0, c00);
    c01 = wmma_f16(af0, bf1, c01);
    c10 = wmma_f16(af1, bf0, c10);
    c11 = wmma_f16(af1, bf1, c11);

    af0 = an0; af1 = an1;
  }

  // ---- epilogue: wave owns rows m0..m0+31, cols n0..n0+31 ------------------
  int nL = lane & 15, mOff = (lane >> 4) * 8;
  long cz = (long)z * cStride;
  float bias0 = bz[n0 + nL], bias1 = bz[n0 + 16 + nL];
  v8f rc[2][2] = {{c00, c01}, {c10, c11}};

#pragma unroll
  for (int rt = 0; rt < 2; rt++) {
    int mb = m0 + rt * 16;
    if (mode == 3) {                       // transposed f16 store, ld = M
      v8h o0, o1;
#pragma unroll
      for (int r = 0; r < 8; r++) {
        o0[r] = (_Float16)(rc[rt][0][r] + bias0);
        o1[r] = (_Float16)(rc[rt][1][r] + bias1);
      }
      *(v8h*)(Ch + cz + (long)(n0 + nL)      * M + mb + mOff) = o0;
      *(v8h*)(Ch + cz + (long)(n0 + 16 + nL) * M + mb + mOff) = o1;
      continue;
    }
#pragma unroll
    for (int r = 0; r < 8; r++) {
      int m   = mb + mOff + r;
      long i0 = cz + (long)m * Nc + n0 + nL;
      long i1 = i0 + 16;
      float v0 = rc[rt][0][r] + bias0;
      float v1 = rc[rt][1][r] + bias1;
      if (mode == 1) {
        v0 += resScale[n0 + nL]      * res[i0];
        v1 += resScale[n0 + 16 + nL] * res[i1];
      }
      if (mode == 4) { v0 = gelu_exact(v0); v1 = gelu_exact(v1); }
      if (mode == 2 || mode == 4) { Ch[i0] = (_Float16)v0; Ch[i1] = (_Float16)v1; }
      else                        { Cf[i0] = v0;            Cf[i1] = v1; }
    }
  }
}

// ---------------------------------------------------------------------------
// Fused sigmoid attention: O[m, c] = sum_k sigmoid(Q.K^T/sqrt(DK)) * V[k, c]
// grid = (B*HEADS, N/128, DIM/96); block = 8 waves, each: 16 rows x 96 cols.
// Shared K tile (32x96 halves) and V^T tile (96x32 halves) are double-buffered
// in LDS via async copies (3 x b128 per thread per step); Q stays in registers.
// S accum -> A-fragment relayout uses a per-wave private LDS tile.
// ---------------------------------------------------------------------------
__global__ __launch_bounds__(256)
void attn_sigmoid_kernel(const _Float16* __restrict__ qh, const _Float16* __restrict__ kh,
                         const _Float16* __restrict__ vhT, _Float16* __restrict__ ocat)
{
  __shared__ _Float16 sKV[2][6144];      // [K tile 32x96 | V tile 96x32], 2x12KB
  __shared__ _Float16 sS[8][16 * 32];    // per-wave sigmoid relayout tiles

  int tid  = threadIdx.x;
  int lane = tid & 31;
  int wv   = tid >> 5;
  int bh   = blockIdx.x;
  int h    = bh >> 3;
  int b    = bh & 7;
  int m0   = blockIdx.y * 128 + wv * 16;
  int cb   = blockIdx.z * 96;

  const _Float16* qp = qh  + (long)bh * Nc_ * DKc + (long)m0 * DKc;
  const _Float16* kp = kh  + (long)bh * Nc_ * DKc;
  const _Float16* vp = vhT + (long)bh * DIMc * Nc_ + (long)cb * Nc_;

  // staging assignments: 768 16B chunks per step (K: 384, V: 384), 3/thread
  const _Float16* gbase[3];
  long gstep[3];
  unsigned ldst[2][3];
#pragma unroll
  for (int j = 0; j < 3; j++) {
    int cid = tid + j * 256;
    int lofs;
    if (cid < 384) {                     // K tile: 32 keys x 96 halves
      int row = cid / 12, cc = cid % 12;
      gbase[j] = kp + (long)row * DKc + cc * 8;
      gstep[j] = 32 * DKc;
      lofs = row * DKc + cc * 8;
    } else {                             // V tile: 96 channels x 32 halves
      int c2 = cid - 384;
      int row = c2 >> 2, cc = c2 & 3;
      gbase[j] = vp + (long)row * Nc_ + cc * 8;
      gstep[j] = 32;
      lofs = 3072 + row * 32 + cc * 8;
    }
    ldst[0][j] = (unsigned)(uintptr_t)&sKV[0][lofs];
    ldst[1][j] = (unsigned)(uintptr_t)&sKV[1][lofs];
  }

  v16h qf[3];
#pragma unroll
  for (int kc = 0; kc < 3; kc++) qf[kc] = load_frag_a(qp + kc * 32, DKc, lane);

  v8f acc[6];
#pragma unroll
  for (int cf = 0; cf < 6; cf++) acc[cf] = (v8f){};

#pragma unroll
  for (int j = 0; j < 3; j++) async_copy_b128(ldst[0][j], gbase[j]);

  const float scale = 0.1020620726f;     // 1/sqrt(96)
  int nL = lane & 15, mOff = (lane >> 4) * 8;
  int mA = lane & 15, koA = (lane >> 4) * 8;
  _Float16* myS = &sS[wv][0];
  int steps = Nc_ / 32;                  // 32 key-steps

  for (int i = 0; i < steps; i++) {
    wait_async0();                       // K/V tiles for buffer i&1 landed
    __syncthreads();                     // publish; buffer (i+1)&1 reusable
    if (i + 1 < steps)
#pragma unroll
      for (int j = 0; j < 3; j++)
        async_copy_b128(ldst[(i + 1) & 1][j], gbase[j] + (long)(i + 1) * gstep[j]);

    const _Float16* kb_ = &sKV[i & 1][0];
    const _Float16* vb_ = kb_ + 3072;

    // S tile (16 rows x 32 keys) over DK=96 channels, K frags from LDS
    v8f s0 = {}, s1 = {};
#pragma unroll
    for (int kc = 0; kc < 3; kc++) {
      s0 = wmma_f16(qf[kc], frag_b_ld(kb_ + kc * 32,           DKc, lane), s0);
      s1 = wmma_f16(qf[kc], frag_b_ld(kb_ + 16 * DKc + kc * 32, DKc, lane), s1);
    }
    // sigmoid + relayout accum -> A-fragment via per-wave LDS tile
#pragma unroll
    for (int r = 0; r < 8; r++) {
      int row = r + mOff;
      myS[row * 32 + nL]      = (_Float16)(1.0f / (1.0f + __expf(-s0[r] * scale)));
      myS[row * 32 + nL + 16] = (_Float16)(1.0f / (1.0f + __expf(-s1[r] * scale)));
    }
    v8h lo = *(const v8h*)&myS[mA * 32 + koA];
    v8h hi = *(const v8h*)&myS[mA * 32 + koA + 16];
    v16h sf;
#pragma unroll
    for (int t = 0; t < 8; t++) { sf[t] = lo[t]; sf[8 + t] = hi[t]; }

    // O += S * V, V frags from LDS
#pragma unroll
    for (int cf = 0; cf < 6; cf++)
      acc[cf] = wmma_f16(sf, frag_b_ld(vb_ + cf * 16 * 32, 32, lane), acc[cf]);
  }

  _Float16* op = ocat + ((long)b * Nc_ + m0) * (HEADSc * DIMc) + h * DIMc + cb;
#pragma unroll
  for (int cf = 0; cf < 6; cf++)
#pragma unroll
    for (int r = 0; r < 8; r++)
      op[(long)(mOff + r) * (HEADSc * DIMc) + cf * 16 + nL] = (_Float16)acc[cf][r];
}

// ---------------------------------------------------------------------------
// Elementwise / reduction kernels
// ---------------------------------------------------------------------------
__global__ __launch_bounds__(256)
void nchw_to_bnc_kernel(const float* __restrict__ dec, float* __restrict__ xf,
                        _Float16* __restrict__ xh)
{
  long i = (long)blockIdx.x * 256 + threadIdx.x;
  if (i >= (long)Bc * Nc_ * DIMc) return;
  int c = (int)(i % DIMc);
  long bn = i / DIMc;
  int n = (int)(bn % Nc_);
  int b = (int)(bn / Nc_);
  float v = dec[((long)b * DIMc + c) * Nc_ + n];
  xf[i] = v;
  xh[i] = (_Float16)v;
}

__global__ __launch_bounds__(256)
void transpose_w_kernel(const float* __restrict__ src, _Float16* __restrict__ dst,
                        int rows, int cols)
{
  long base = (long)blockIdx.z * rows * cols;
  long i = (long)blockIdx.x * 256 + threadIdx.x;
  if (i >= (long)rows * cols) return;
  int c = (int)(i % cols);
  int r = (int)(i / cols);
  dst[base + (long)c * rows + r] = (_Float16)src[base + i];
}

__global__ __launch_bounds__(256)
void peg_kernel(const float* __restrict__ x, const float* __restrict__ w,
                const float* __restrict__ bias, float* __restrict__ y)
{
  long i = (long)blockIdx.x * 256 + threadIdx.x;
  if (i >= (long)Bc * Nc_ * DIMc) return;
  int c = (int)(i % DIMc);
  long bn = i / DIMc;
  int n = (int)(bn % Nc_);
  int b = (int)(bn / Nc_);
  int hh = n / Wc, ww = n % Wc;
  float acc = bias[c];
#pragma unroll
  for (int dy = -1; dy <= 1; dy++)
#pragma unroll
    for (int dx = -1; dx <= 1; dx++) {
      int h2 = hh + dy, w2 = ww + dx;
      if (h2 >= 0 && h2 < Hc && w2 >= 0 && w2 < Wc)
        acc += w[c * 9 + (dy + 1) * 3 + (dx + 1)] *
               x[((long)b * Nc_ + h2 * Wc + w2) * DIMc + c];
    }
  y[i] = acc;
}

// one wave (32 lanes) per row of 384; outputs optional f32 / f16 / NCHW-f32
__global__ __launch_bounds__(256)
void ln_kernel(const float* __restrict__ x, const float* __restrict__ g,
               const float* __restrict__ bta, float* __restrict__ of,
               _Float16* __restrict__ oh, float* __restrict__ onchw)
{
  int lane = threadIdx.x & 31;
  int wv   = threadIdx.x >> 5;
  long row = (long)blockIdx.x * 8 + wv;
  const float* xr = x + row * DIMc;
  float vals[12];
  float s = 0.f, q = 0.f;
#pragma unroll
  for (int i = 0; i < 12; i++) {
    float v = xr[lane + i * 32];
    vals[i] = v; s += v; q += v * v;
  }
#pragma unroll
  for (int off = 16; off > 0; off >>= 1) {
    s += __shfl_xor(s, off, 32);
    q += __shfl_xor(q, off, 32);
  }
  float mu = s * (1.0f / DIMc);
  float var = q * (1.0f / DIMc) - mu * mu;
  float rs = rsqrtf(var + EPSc);
#pragma unroll
  for (int i = 0; i < 12; i++) {
    int c = lane + i * 32;
    float o = (vals[i] - mu) * rs * g[c] + bta[c];
    if (of)    of[row * DIMc + c] = o;
    if (oh)    oh[row * DIMc + c] = (_Float16)o;
    if (onchw) {
      int b = (int)(row / Nc_);
      int n = (int)(row % Nc_);
      onchw[((long)b * DIMc + c) * Nc_ + n] = o;
    }
  }
}

// ---------------------------------------------------------------------------
// Host-side orchestration
// ---------------------------------------------------------------------------
extern "C" void kernel_launch(void* const* d_in, const int* in_sizes, int n_in,
                              void* d_out, int out_size, void* d_ws, size_t ws_size,
                              hipStream_t stream)
{
  (void)in_sizes; (void)n_in; (void)out_size; (void)ws_size;
  const float* dec   = (const float*)d_in[0];
  const float* wq[2] = {(const float*)d_in[1],  (const float*)d_in[9]};
  const float* bq[2] = {(const float*)d_in[2],  (const float*)d_in[10]};
  const float* wk[2] = {(const float*)d_in[3],  (const float*)d_in[11]};
  const float* bk[2] = {(const float*)d_in[4],  (const float*)d_in[12]};
  const float* wvv[2]= {(const float*)d_in[5],  (const float*)d_in[13]};
  const float* bv[2] = {(const float*)d_in[6],  (const float*)d_in[14]};
  const float* wp[2] = {(const float*)d_in[7],  (const float*)d_in[15]};
  const float* bp[2] = {(const float*)d_in[8],  (const float*)d_in[16]};
  const float* peg_w = (const float*)d_in[17];
  const float* peg_b = (const float*)d_in[18];
  const float* ln1g  = (const float*)d_in[19];
  const float* ln1b  = (const float*)d_in[20];
  const float* mlng  = (const float*)d_in[21];
  const float* mlnb  = (const float*)d_in[22];
  const float* mw1   = (const float*)d_in[23];
  const float* mb1   = (const float*)d_in[24];
  const float* mw2   = (const float*)d_in[25];
  const float* mb2   = (const float*)d_in[26];
  const float* ln2g  = (const float*)d_in[27];
  const float* ln2b  = (const float*)d_in[28];
  const float* a1    = (const float*)d_in[29];
  const float* a2    = (const float*)d_in[30];
  const float* a3    = (const float*)d_in[31];

  const long BN = (long)Bc * Nc_;                 // 8192 rows
  char* base = (char*)d_ws;
  size_t off = 0;
  auto alloc = [&](size_t bytes) -> void* {
    off = (off + 255) & ~(size_t)255;
    void* p = base + off;
    off += bytes;
    return p;
  };

  float*    xf32  = (float*)   alloc(BN * DIMc * 4);
  _Float16* xf16  = (_Float16*)alloc(BN * DIMc * 2);
  float*    x1f32 = (float*)   alloc(BN * DIMc * 4);
  float*    x2f32 = (float*)   alloc(BN * DIMc * 4);
  _Float16* x2f16 = (_Float16*)alloc(BN * DIMc * 2);
  float*    x3f32 = (float*)   alloc(BN * DIMc * 4);
  _Float16* qhB   = (_Float16*)alloc((long)64 * Nc_ * DKc * 2);
  _Float16* khB   = (_Float16*)alloc((long)64 * Nc_ * DKc * 2);
  _Float16* vhT   = (_Float16*)alloc((long)64 * DIMc * Nc_ * 2);
  _Float16* ocat  = (_Float16*)alloc(BN * (long)(HEADSc * DIMc) * 2);
  _Float16* wqT[2], *wkT[2], *wvT[2], *wpT[2];
  for (int i = 0; i < 2; i++) {
    wqT[i] = (_Float16*)alloc((long)HEADSc * DKc  * DIMc * 2);
    wkT[i] = (_Float16*)alloc((long)HEADSc * DKc  * DIMc * 2);
    wvT[i] = (_Float16*)alloc((long)HEADSc * DIMc * DIMc * 2);
    wpT[i] = (_Float16*)alloc((long)(HEADSc * DIMc) * DIMc * 2);
  }
  _Float16* w1T = (_Float16*)alloc((long)DIMc * HID * 2);
  _Float16* w2T = (_Float16*)alloc((long)HID * DIMc * 2);
  // buffer reuse (lifetimes are disjoint):
  float*    pegout = xf32;            // xf32 dead after wp1 residual
  _Float16* hln    = xf16;            // xf16 dead after block-1 q/k/v
  _Float16* hid    = qhB;             // qh dead after attention-2 (same 12.58MB)
  float*    x4f32  = x1f32;           // x1 dead after PEG

  const int ew = 256;
  auto cdiv = [](long a, long b) { return (int)((a + b - 1) / b); };

  // ---- weight transpose+convert to f16 -------------------------------------
  for (int i = 0; i < 2; i++) {
    transpose_w_kernel<<<dim3(cdiv((long)DIMc * DKc, ew), 1, HEADSc), ew, 0, stream>>>(wq[i],  wqT[i], DIMc, DKc);
    transpose_w_kernel<<<dim3(cdiv((long)DIMc * DKc, ew), 1, HEADSc), ew, 0, stream>>>(wk[i],  wkT[i], DIMc, DKc);
    transpose_w_kernel<<<dim3(cdiv((long)DIMc * DIMc, ew), 1, HEADSc), ew, 0, stream>>>(wvv[i], wvT[i], DIMc, DIMc);
    transpose_w_kernel<<<dim3(cdiv((long)HEADSc * DIMc * DIMc, ew), 1, 1), ew, 0, stream>>>(wp[i], wpT[i], HEADSc * DIMc, DIMc);
  }
  transpose_w_kernel<<<dim3(cdiv((long)DIMc * HID, ew), 1, 1), ew, 0, stream>>>(mw1, w1T, DIMc, HID);
  transpose_w_kernel<<<dim3(cdiv((long)HID * DIMc, ew), 1, 1), ew, 0, stream>>>(mw2, w2T, HID, DIMc);

  // ---- NCHW -> [B,N,C] (f32 + f16) -----------------------------------------
  nchw_to_bnc_kernel<<<cdiv(BN * DIMc, ew), ew, 0, stream>>>(dec, xf32, xf16);

  // ---- transformer blocks ---------------------------------------------------
  auto run_block = [&](const _Float16* xh, const float* xres, const float* alpha,
                       int i, float* attn_out) {
    // q/k projections: z = h*8+b, M=1024, Nc=96, K=384
    gemm_wmma_kernel<<<dim3(3, 4, 64), 256, 0, stream>>>(
        xh, wqT[i], bq[i], nullptr, qhB, nullptr, nullptr,
        Nc_, DKc, DIMc, 8, (long)Nc_ * DIMc, 8, (long)DKc * DIMc,
        (long)Nc_ * DKc, DKc, 2);
    gemm_wmma_kernel<<<dim3(3, 4, 64), 256, 0, stream>>>(
        xh, wkT[i], bk[i], nullptr, khB, nullptr, nullptr,
        Nc_, DKc, DIMc, 8, (long)Nc_ * DIMc, 8, (long)DKc * DIMc,
        (long)Nc_ * DKc, DKc, 2);
    // v projection with transposed f16 store -> V^T [bh][DIM][N]
    gemm_wmma_kernel<<<dim3(12, 4, 64), 256, 0, stream>>>(
        xh, wvT[i], bv[i], nullptr, vhT, nullptr, nullptr,
        Nc_, DIMc, DIMc, 8, (long)Nc_ * DIMc, 8, (long)DIMc * DIMc,
        (long)DIMc * Nc_, DIMc, 3);
    // fused sigmoid attention -> ocat f16
    attn_sigmoid_kernel<<<dim3(64, 8, 4), 256, 0, stream>>>(qhB, khB, vhT, ocat);
    // output projection + alpha*x residual -> f32
    gemm_wmma_kernel<<<dim3(12, 4, 8), 256, 0, stream>>>(
        ocat, wpT[i], bp[i], attn_out, nullptr, xres, alpha,
        Nc_, DIMc, HEADSc * DIMc, 8, (long)Nc_ * HEADSc * DIMc, 1, 0,
        (long)Nc_ * DIMc, 0, 1);
  };

  // block 1
  run_block(xf16, xf32, a1, 0, x1f32);
  peg_kernel<<<cdiv(BN * DIMc, ew), ew, 0, stream>>>(x1f32, peg_w, peg_b, pegout);
  ln_kernel<<<(int)(BN / 8), 256, 0, stream>>>(pegout, ln1g, ln1b, x2f32, x2f16, nullptr);

  // block 2
  run_block(x2f16, x2f32, a2, 1, x3f32);

  // MLP
  ln_kernel<<<(int)(BN / 8), 256, 0, stream>>>(x3f32, mlng, mlnb, nullptr, hln, nullptr);
  gemm_wmma_kernel<<<dim3(24, 32, 1), 256, 0, stream>>>(        // 8192x768x384 + GELU
      hln, w1T, mb1, nullptr, hid, nullptr, nullptr,
      (int)BN, HID, DIMc, 1, 0, 1, 0, 0, 0, 4);
  gemm_wmma_kernel<<<dim3(12, 32, 1), 256, 0, stream>>>(        // 8192x384x768 + a3*x residual
      hid, w2T, mb2, x4f32, nullptr, x3f32, a3,
      (int)BN, DIMc, HID, 1, 0, 1, 0, 0, 0, 1);

  // final LayerNorm, written straight to NCHW f32 output
  ln_kernel<<<(int)(BN / 8), 256, 0, stream>>>(x4f32, ln2g, ln2b, nullptr, nullptr,
                                               (float*)d_out);
}
